// FQ_LISCS_Decoder_52158082843117
// MI455X (gfx1250) — matrix operand order repaired and checked by reading, not compiled, not run on hardware
//
#include <hip/hip_runtime.h>

// ---------------------------------------------------------------------------
// FQ-LISCS Swin decoder forward for gfx1250 (MI455X).
// All matmuls lower to v_wmma_f32_16x16x32_f16 (wave32 WMMA, f32 accumulate).
// Epilogues (bias / SNR mod-scale / sigmoid gate / GELU / residual / pixel-
// shuffle upsample / NCHW output) fused into the GEMM. Window attention fully
// fused in LDS per (batch, window, head). All fragment loads are branch-free
// contiguous loads (zero-padded LDS tiles, clamped B columns).
// ---------------------------------------------------------------------------

typedef __attribute__((ext_vector_type(16))) _Float16 v16h;
typedef __attribute__((ext_vector_type(8)))  float    v8f;

__device__ __forceinline__ v8f wmma_f16(v16h a, v16h b, v8f c) {
  // 8 args: (neg_a, A, neg_b, B, c_mod, C, reuse_a, reuse_b)
  return __builtin_amdgcn_wmma_f32_16x16x32_f16(false, a, false, b, (short)0, c,
                                                false, false);
}

enum { EP_NONE = 0, EP_MODSCALE, EP_GATE, EP_RESADD, EP_GELU, EP_UP, EP_UPOUT };

// ---------------------------------------------------------------------------
// Generic WMMA GEMM: out[M,N] = epilogue(A[M,K] @ Wt[N,K]^T + bias)
// A: f32 row-major. Wt: f16, transposed (N-major) so B-frag loads are
// contiguous. Block = 128 threads = 4 waves; each wave owns a 16x64 strip.
// M % 64 == 0 and K % 32 == 0 always hold for this model. Out-of-range N
// columns are CLAMPED (not predicated): a clamped B column only corrupts its
// own output column, which is never stored.
// ---------------------------------------------------------------------------
template <int EP>
__global__ __launch_bounds__(128) void gemm_kernel(
    const float* __restrict__ A, const _Float16* __restrict__ Wt,
    const float* __restrict__ bias, float* __restrict__ out,
    const float* __restrict__ extra, int M, int N, int K,
    int P0, int P1, int P2) {
  const int lane = threadIdx.x & 31;
  const int wave = threadIdx.x >> 5;
  const int lm = lane & 15;
  const int hi = lane >> 4;  // 0 or 1 (half-wave)
  const int m0 = blockIdx.y * 64 + wave * 16;
  const int n0 = blockIdx.x * 64;

  v8f acc[4] = {};
  const float* arow = A + (size_t)(m0 + lm) * K;

  // Hoisted, clamped B base pointers (branch-free fragment loads).
  const _Float16* wbase[4];
#pragma unroll
  for (int nt = 0; nt < 4; ++nt) {
    int col = n0 + nt * 16 + lm;
    if (col > N - 1) col = N - 1;
    wbase[nt] = Wt + (size_t)col * K + hi * 16;
  }

  for (int k0 = 0; k0 < K; k0 += 32) {
    // A fragment (16x32 f16): lane row = m0+lm, contiguous K order k = 16*hi + e (same as B).
    v16h a;
#pragma unroll
    for (int e = 0; e < 8; ++e) a[e] = (_Float16)arow[k0 + hi * 16 + e];
#pragma unroll
    for (int e = 0; e < 8; ++e) a[8 + e] = (_Float16)arow[k0 + hi * 16 + 8 + e];
#pragma unroll
    for (int nt = 0; nt < 4; ++nt) {
      const _Float16* wr = wbase[nt] + k0;
      v16h b;
#pragma unroll
      for (int e = 0; e < 16; ++e) b[e] = wr[e];
      acc[nt] = wmma_f16(a, b, acc[nt]);
    }
  }

#pragma unroll
  for (int nt = 0; nt < 4; ++nt) {
    const int n = n0 + nt * 16 + lm;
    if (n >= N) continue;
    const float bv = bias ? bias[n] : 0.f;
#pragma unroll
    for (int r = 0; r < 8; ++r) {
      const int m = m0 + hi * 8 + r;
      float v = acc[nt][r] + bv;
      if (EP == EP_MODSCALE) {
        v *= extra[n];  // per-output-channel SNR modulation
      } else if (EP == EP_GATE) {
        float aa = fminf(fmaxf(v, -8.f), 8.f);
        v = extra[(size_t)m * N + n] * (1.f / (1.f + expf(-aa)));
      } else if (EP == EP_RESADD) {
        v += extra[(size_t)m * N + n];
      } else if (EP == EP_GELU) {
        v = 0.5f * v * (1.f + erff(v * 0.70710678118654752f));
      }
      if (EP == EP_UP) {  // PatchReverseMerging scatter: (B,L,4*od)->(B,4L,od)
        const int b = m / P0, l = m - b * P0;          // P0 = L
        const int hh = l / P1, wwp = l - hh * P1;       // P1 = W
        const int r2 = n / (2 * P2);                    // P2 = od
        const int rem = n - r2 * (2 * P2);
        const int c2 = rem / P2, co = rem - c2 * P2;
        const int row2 = b * (4 * P0) + (hh * 2 + r2) * (2 * P1) + (wwp * 2 + c2);
        out[(size_t)row2 * P2 + co] = v;
      } else if (EP == EP_UPOUT) {  // same + final (B,3,256,256) NCHW store
        const int b = m / P0, l = m - b * P0;
        const int hh = l / P1, wwp = l - hh * P1;
        const int r2 = n / (2 * P2);
        const int rem = n - r2 * (2 * P2);
        const int c2 = rem / P2, co = rem - c2 * P2;
        const int HW = 2 * P1;  // 256
        out[(((size_t)(b * 3 + co) * HW) + (hh * 2 + r2)) * HW + (wwp * 2 + c2)] = v;
      } else {
        out[(size_t)m * N + n] = v;
      }
    }
  }
}

// ---------------------------------------------------------------------------
// LayerNorm over the channel dim (C = 64 or 128), one token per thread.
// ---------------------------------------------------------------------------
__global__ void ln_kernel(const float* __restrict__ x, const float* __restrict__ g,
                          const float* __restrict__ b, float* __restrict__ o,
                          int M, int C) {
  const int row = blockIdx.x * blockDim.x + threadIdx.x;
  if (row >= M) return;
  const float* xr = x + (size_t)row * C;
  float mu = 0.f;
  for (int c = 0; c < C; ++c) mu += xr[c];
  mu /= (float)C;
  float var = 0.f;
  for (int c = 0; c < C; ++c) { float d = xr[c] - mu; var += d * d; }
  var /= (float)C;
  const float inv = rsqrtf(var + 1e-5f);
  float* orow = o + (size_t)row * C;
  for (int c = 0; c < C; ++c) orow[c] = (xr[c] - mu) * inv * g[c] + b[c];
}

// ---------------------------------------------------------------------------
// SNR modulation vector: m = sigmoid(relu(snr*w1 + b1) @ w2 + b2), 192-wide.
// Depends only on the scalar snr -> one 192-thread block.
// ---------------------------------------------------------------------------
__global__ void modvec_kernel(const float* __restrict__ snr,
                              const float* __restrict__ w1, const float* __restrict__ b1,
                              const float* __restrict__ w2, const float* __restrict__ b2,
                              float* __restrict__ mvec) {
  __shared__ float h[192];
  const int j = threadIdx.x;
  const float s = snr[0];
  float t = s * w1[j] + b1[j];
  h[j] = t > 0.f ? t : 0.f;
  __syncthreads();
  float acc = b2[j];
  for (int k = 0; k < 192; ++k) acc += h[k] * w2[k * 192 + j];
  mvec[j] = 1.f / (1.f + expf(-acc));
}

// ---------------------------------------------------------------------------
// Weight convert+transpose: W[K,N] f32 -> Wt[N,K] f16.
// ---------------------------------------------------------------------------
__global__ void wconv_kernel(const float* __restrict__ W, _Float16* __restrict__ Wt,
                             int K, int N) {
  const int i = blockIdx.x * 256 + threadIdx.x;
  if (i >= K * N) return;
  const int k = i / N, n = i - k * N;
  Wt[(size_t)n * K + k] = (_Float16)W[i];
}

// ---------------------------------------------------------------------------
// Fused window attention. One block (128 thr = 4 waves) per (window, head, b).
// ws=8 -> 64 tokens, hd=16. Q.K^T via WMMA (zero-padded K=32 LDS tiles ->
// branch-free b128 fragment loads), bias + shift-mask + softmax in LDS,
// P.V via WMMA (K=64, V transposed at staging for contiguous B-frags).
// Shift handled by gather/scatter with rolled indices; mask regions computed
// analytically (matches _shift_mask label construction).
// ---------------------------------------------------------------------------
__global__ __launch_bounds__(128) void attn_kernel(
    const float* __restrict__ qkv, const float* __restrict__ rel_bias,
    float* __restrict__ out, int Hdim, int C, int heads, int shift) {
  __shared__ _Float16 qs[64 * 32];  // 64 rows x K=32 (upper 16 zero)
  __shared__ _Float16 ks[64 * 32];  // 64 cols x K=32 (upper 16 zero)
  __shared__ _Float16 vt[16 * 64];  // transposed V: [d][token]
  __shared__ float S[64 * 64];
  __shared__ _Float16 P[64 * 64];

  const int nW = Hdim >> 3;
  const int wh = blockIdx.x / nW, ww = blockIdx.x % nW;
  const int head = blockIdx.y;
  const int b = blockIdx.z;
  const int L = Hdim * Hdim;
  const int C3 = 3 * C;
  const int tid = threadIdx.x;

  // Stage q/k into zero-padded 64x32 tiles; v transposed into vt[16][64].
  for (int idx = tid; idx < 64 * 32; idx += 128) {
    const int t = idx >> 5, kk = idx & 31;
    _Float16 qv = (_Float16)0.f, kv = (_Float16)0.f;
    if (kk < 16) {
      const int i = t >> 3, j = t & 7;
      const int h = (wh * 8 + i + shift) % Hdim;
      const int w = (ww * 8 + j + shift) % Hdim;
      const size_t base = (size_t)(b * L + h * Hdim + w) * C3 + head * 16 + kk;
      qv = (_Float16)qkv[base];
      kv = (_Float16)qkv[base + C];
    }
    qs[idx] = qv;
    ks[idx] = kv;
  }
  for (int idx = tid; idx < 64 * 16; idx += 128) {
    const int t = idx >> 4, d = idx & 15;
    const int i = t >> 3, j = t & 7;
    const int h = (wh * 8 + i + shift) % Hdim;
    const int w = (ww * 8 + j + shift) % Hdim;
    vt[d * 64 + t] = (_Float16)qkv[(size_t)(b * L + h * Hdim + w) * C3 + 2 * C + head * 16 + d];
  }
  __syncthreads();

  const int lane = tid & 31, wave = tid >> 5;
  const int lm = lane & 15, hi = lane >> 4;

  // S = q @ k^T  (64x64, K=16 zero-padded to 32). Wave w owns row-block w.
  {
    const int row = wave * 16 + lm;
    v16h a;
#pragma unroll
    for (int e = 0; e < 8; ++e) a[e] = qs[row * 32 + hi * 16 + e];
#pragma unroll
    for (int e = 0; e < 8; ++e) a[8 + e] = qs[row * 32 + hi * 16 + 8 + e];
#pragma unroll
    for (int nt = 0; nt < 4; ++nt) {
      const int col = nt * 16 + lm;
      v16h bf;
#pragma unroll
      for (int e = 0; e < 16; ++e) bf[e] = ks[col * 32 + hi * 16 + e];
      v8f c = {};
      c = wmma_f16(a, bf, c);
#pragma unroll
      for (int r = 0; r < 8; ++r)
        S[(wave * 16 + hi * 8 + r) * 64 + nt * 16 + lm] = c[r];
    }
  }
  __syncthreads();

  // scale + rel-pos bias + shift mask + softmax, one row per thread (tid<64).
  if (tid < 64) {
    const int q = tid, qi = q >> 3, qj = q & 7;
    int labq = 0;
    if (shift) {
      const int h = wh * 8 + qi, w = ww * 8 + qj;
      const int rh = (h < Hdim - 8) ? 0 : ((h < Hdim - shift) ? 1 : 2);
      const int rw = (w < Hdim - 8) ? 0 : ((w < Hdim - shift) ? 1 : 2);
      labq = rh * 3 + rw;
    }
    float mx = -1e30f;
    for (int k = 0; k < 64; ++k) {
      const int ki = k >> 3, kj = k & 7;
      const int ridx = (qi - ki + 7) * 15 + (qj - kj + 7);
      float s = S[q * 64 + k] * 0.25f + rel_bias[ridx * heads + head];
      if (shift) {
        const int h = wh * 8 + ki, w = ww * 8 + kj;
        const int rh = (h < Hdim - 8) ? 0 : ((h < Hdim - shift) ? 1 : 2);
        const int rw = (w < Hdim - 8) ? 0 : ((w < Hdim - shift) ? 1 : 2);
        if (rh * 3 + rw != labq) s -= 100.f;
      }
      S[q * 64 + k] = s;
      mx = fmaxf(mx, s);
    }
    float sum = 0.f;
    for (int k = 0; k < 64; ++k) {
      const float e = expf(S[q * 64 + k] - mx);
      S[q * 64 + k] = e;
      sum += e;
    }
    const float inv = 1.f / sum;
    for (int k = 0; k < 64; ++k) P[q * 64 + k] = (_Float16)(S[q * 64 + k] * inv);
  }
  __syncthreads();

  // O = P @ v  (64x16, K=64 in two 32-steps), then scatter back (roll undo).
  {
    const int row = wave * 16 + lm;
    v8f c = {};
#pragma unroll
    for (int k0 = 0; k0 < 64; k0 += 32) {
      v16h a, bf;
#pragma unroll
      for (int e = 0; e < 8; ++e) a[e] = P[row * 64 + k0 + hi * 16 + e];
#pragma unroll
      for (int e = 0; e < 8; ++e) a[8 + e] = P[row * 64 + k0 + hi * 16 + 8 + e];
#pragma unroll
      for (int e = 0; e < 16; ++e) bf[e] = vt[lm * 64 + k0 + hi * 16 + e];
      c = wmma_f16(a, bf, c);
    }
#pragma unroll
    for (int r = 0; r < 8; ++r) {
      const int t = wave * 16 + hi * 8 + r;
      const int i = t >> 3, j = t & 7;
      const int h = (wh * 8 + i + shift) % Hdim;
      const int w = (ww * 8 + j + shift) % Hdim;
      out[(size_t)(b * L + h * Hdim + w) * C + head * 16 + lm] = c[r];
    }
  }
}

// ---------------------------------------------------------------------------
// Host orchestration.
// ---------------------------------------------------------------------------
static inline void launch_gemm(int ep, const float* A, const _Float16* Wt,
                               const float* bias, float* out, const float* extra,
                               int M, int N, int K, int P0, int P1, int P2,
                               hipStream_t s) {
  dim3 grid((N + 63) / 64, M / 64), blk(128);
  switch (ep) {
    case EP_NONE:     gemm_kernel<EP_NONE><<<grid, blk, 0, s>>>(A, Wt, bias, out, extra, M, N, K, P0, P1, P2); break;
    case EP_MODSCALE: gemm_kernel<EP_MODSCALE><<<grid, blk, 0, s>>>(A, Wt, bias, out, extra, M, N, K, P0, P1, P2); break;
    case EP_GATE:     gemm_kernel<EP_GATE><<<grid, blk, 0, s>>>(A, Wt, bias, out, extra, M, N, K, P0, P1, P2); break;
    case EP_RESADD:   gemm_kernel<EP_RESADD><<<grid, blk, 0, s>>>(A, Wt, bias, out, extra, M, N, K, P0, P1, P2); break;
    case EP_GELU:     gemm_kernel<EP_GELU><<<grid, blk, 0, s>>>(A, Wt, bias, out, extra, M, N, K, P0, P1, P2); break;
    case EP_UP:       gemm_kernel<EP_UP><<<grid, blk, 0, s>>>(A, Wt, bias, out, extra, M, N, K, P0, P1, P2); break;
    case EP_UPOUT:    gemm_kernel<EP_UPOUT><<<grid, blk, 0, s>>>(A, Wt, bias, out, extra, M, N, K, P0, P1, P2); break;
  }
}

extern "C" void kernel_launch(void* const* d_in, const int* in_sizes, int n_in,
                              void* d_out, int out_size, void* d_ws, size_t ws_size,
                              hipStream_t stream) {
  (void)in_sizes; (void)n_in; (void)out_size; (void)ws_size;

  // ---- input pointer map (setup_inputs dict insertion order, recursive) ----
  const float* X_IN = (const float*)d_in[0];   // (4, 4096, 32)
  const float* SNR  = (const float*)d_in[1];   // (1,)
  int p = 2;
  const float* head_w = (const float*)d_in[p++];
  const float* head_b = (const float*)d_in[p++];
  const float *sm_w[8], *sm_b[8];
  for (int i = 0; i < 8; ++i) { sm_w[i] = (const float*)d_in[p++]; sm_b[i] = (const float*)d_in[p++]; }
  const float *bm_w1[7], *bm_b1[7], *bm_w2[7], *bm_b2[7];
  for (int i = 0; i < 7; ++i) {
    bm_w1[i] = (const float*)d_in[p++]; bm_b1[i] = (const float*)d_in[p++];
    bm_w2[i] = (const float*)d_in[p++]; bm_b2[i] = (const float*)d_in[p++];
  }
  struct Blk { const float *ln1_g,*ln1_b,*qkv_w,*qkv_b,*proj_w,*proj_b,*rel,*ln2_g,*ln2_b,*fc1_w,*fc1_b,*fc2_w,*fc2_b; };
  struct Lay { Blk blk[2]; const float *up_g,*up_b,*inc_w; };
  Lay lay[2];
  for (int li = 0; li < 2; ++li) {
    for (int bi = 0; bi < 2; ++bi) {
      Blk& k = lay[li].blk[bi];
      k.ln1_g = (const float*)d_in[p++]; k.ln1_b = (const float*)d_in[p++];
      k.qkv_w = (const float*)d_in[p++]; k.qkv_b = (const float*)d_in[p++];
      k.proj_w = (const float*)d_in[p++]; k.proj_b = (const float*)d_in[p++];
      k.rel = (const float*)d_in[p++];
      k.ln2_g = (const float*)d_in[p++]; k.ln2_b = (const float*)d_in[p++];
      k.fc1_w = (const float*)d_in[p++]; k.fc1_b = (const float*)d_in[p++];
      k.fc2_w = (const float*)d_in[p++]; k.fc2_b = (const float*)d_in[p++];
    }
    lay[li].up_g = (const float*)d_in[p++];
    lay[li].up_b = (const float*)d_in[p++];
    lay[li].inc_w = (const float*)d_in[p++];
  }

  // ---- workspace layout (~130 MB, bump-allocated, deterministic) ----
  const size_t MB = 1024u * 1024u;
  char* ws = (char*)d_ws;
  float* Xbuf   = (float*)(ws + 0 * MB);    // 16 MB residual stream A
  float* Ybuf   = (float*)(ws + 16 * MB);   // 16 MB residual stream B
  float* LNbuf  = (float*)(ws + 32 * MB);   // 16 MB LN out / chain ping
  float* ATTbuf = (float*)(ws + 48 * MB);   // 16 MB attn out / chain pong
  float* QKVbuf = (float*)(ws + 64 * MB);   // 64 MB qkv / MLP hidden
  float* MV     = (float*)(ws + 128 * MB);  // 7*192 mod vectors
  _Float16* W16 = (_Float16*)(ws + 128 * MB + 65536);  // ~1.6 MB f16 weights

  size_t w16off = 0;
  auto prep = [&](const float* W, int K, int N) -> _Float16* {
    _Float16* dst = W16 + w16off;
    w16off += (size_t)K * N;
    const int total = K * N;
    wconv_kernel<<<(total + 255) / 256, 256, 0, stream>>>(W, dst, K, N);
    return dst;
  };

  // f16 transposed weights
  _Float16* headT = prep(head_w, 32, 128);
  _Float16* smT[8];
  smT[0] = prep(sm_w[0], 128, 192);
  for (int i = 1; i < 7; ++i) smT[i] = prep(sm_w[i], 192, 192);
  smT[7] = prep(sm_w[7], 192, 128);
  _Float16 *qkvT[2][2], *projT[2][2], *fc1T[2][2], *fc2T[2][2], *incT[2];
  const int Cdim[2] = {128, 64};
  for (int li = 0; li < 2; ++li) {
    const int C = Cdim[li];
    for (int bi = 0; bi < 2; ++bi) {
      qkvT[li][bi] = prep(lay[li].blk[bi].qkv_w, C, 3 * C);
      projT[li][bi] = prep(lay[li].blk[bi].proj_w, C, C);
      fc1T[li][bi] = prep(lay[li].blk[bi].fc1_w, C, 4 * C);
      fc2T[li][bi] = prep(lay[li].blk[bi].fc2_w, 4 * C, C);
    }
    incT[li] = prep(lay[li].inc_w, C, li == 0 ? 256 : 12);
  }

  // SNR modulation vectors (scalar snr -> per-channel scales)
  for (int i = 0; i < 7; ++i)
    modvec_kernel<<<1, 192, 0, stream>>>(SNR, bm_w1[i], bm_b1[i], bm_w2[i], bm_b2[i], MV + i * 192);

  int rows = 4 * 4096;  // B * L0 = 16384

  // Head: X = x @ head_w + head_b   (K=32, N=128)
  launch_gemm(EP_NONE, X_IN, headT, head_b, Xbuf, nullptr, rows, 128, 32, 0, 0, 0, stream);

  // Modulation chain: 7 GEMMs with per-channel scale, then sigmoid gate.
  {
    const float* T_in = Xbuf;
    float* pong[2] = {LNbuf, ATTbuf};
    int Kc = 128;
    for (int i = 0; i < 7; ++i) {
      float* T_out = pong[i & 1];
      launch_gemm(EP_MODSCALE, T_in, smT[i], sm_b[i], T_out, MV + i * 192,
                  rows, 192, Kc, 0, 0, 0, stream);
      T_in = T_out;
      Kc = 192;
    }
    // aa = T6 @ sm7 + b7; X_gated = X * sigmoid(clip(aa, ±8))  -> Ybuf
    launch_gemm(EP_GATE, T_in, smT[7], sm_b[7], Ybuf, Xbuf, rows, 128, 192, 0, 0, 0, stream);
  }

  // Swin decoder: 2 layers x 2 blocks, then upsample.
  float* ptrA = Ybuf;
  float* ptrB = Xbuf;
  int Hd = 64, C = 128, heads = 8, L = 4096;
  for (int li = 0; li < 2; ++li) {
    rows = 4 * L;
    const int nwd = Hd / 8;
    for (int bi = 0; bi < 2; ++bi) {
      const Blk& bk = lay[li].blk[bi];
      const int shift = (bi & 1) ? 4 : 0;
      // LN1 -> qkv
      ln_kernel<<<(rows + 255) / 256, 256, 0, stream>>>(ptrA, bk.ln1_g, bk.ln1_b, LNbuf, rows, C);
      launch_gemm(EP_NONE, LNbuf, qkvT[li][bi], bk.qkv_b, QKVbuf, nullptr, rows, 3 * C, C, 0, 0, 0, stream);
      // fused window attention
      attn_kernel<<<dim3(nwd * nwd, heads, 4), 128, 0, stream>>>(QKVbuf, bk.rel, ATTbuf, Hd, C, heads, shift);
      // proj + residual
      launch_gemm(EP_RESADD, ATTbuf, projT[li][bi], bk.proj_b, ptrB, ptrA, rows, C, C, 0, 0, 0, stream);
      { float* t = ptrA; ptrA = ptrB; ptrB = t; }
      // LN2 -> MLP (GELU) + residual
      ln_kernel<<<(rows + 255) / 256, 256, 0, stream>>>(ptrA, bk.ln2_g, bk.ln2_b, LNbuf, rows, C);
      launch_gemm(EP_GELU, LNbuf, fc1T[li][bi], bk.fc1_b, QKVbuf, nullptr, rows, 4 * C, C, 0, 0, 0, stream);
      launch_gemm(EP_RESADD, QKVbuf, fc2T[li][bi], bk.fc2_b, ptrB, ptrA, rows, C, 4 * C, 0, 0, 0, stream);
      { float* t = ptrA; ptrA = ptrB; ptrB = t; }
    }
    // PatchReverseMerging upsample
    ln_kernel<<<(rows + 255) / 256, 256, 0, stream>>>(ptrA, lay[li].up_g, lay[li].up_b, LNbuf, rows, C);
    if (li == 0) {
      launch_gemm(EP_UP, LNbuf, incT[0], nullptr, ptrB, nullptr, rows, 256, C, L, Hd, 64, stream);
      { float* t = ptrA; ptrA = ptrB; ptrB = t; }
      L *= 4; Hd *= 2; C = 64; heads = 4;
    } else {
      // final: od=3, store directly to d_out as (4, 3, 256, 256)
      launch_gemm(EP_UPOUT, LNbuf, incT[1], nullptr, (float*)d_out, nullptr, rows, 12, C, L, Hd, 3, stream);
    }
  }
}
